// CausalSelfAttention_55576876810363
// MI455X (gfx1250) — matrix operand order repaired and checked by reading, not compiled, not run on hardware
//
#include <hip/hip_runtime.h>

#define B_ 4
#define T_ 2048
#define C_ 1024
#define H_ 16
#define D_ 64

typedef __attribute__((ext_vector_type(16))) __bf16 v16bf;
typedef __attribute__((ext_vector_type(8)))  __bf16 v8bf;
typedef __attribute__((ext_vector_type(8)))  float  v8f;

union Frag { v16bf v; v8bf h[2]; };

static __device__ inline v8f vzero() {
  v8f z = {0.f, 0.f, 0.f, 0.f, 0.f, 0.f, 0.f, 0.f};
  return z;
}

static __device__ inline v8f wmma_bf16(const Frag& a, const Frag& b, v8f c) {
  // D(16x16,f32) = A(16x32,bf16) x B(32x16,bf16) + C
  return __builtin_amdgcn_wmma_f32_16x16x32_bf16(
      /*neg_a=*/false, a.v, /*neg_b=*/false, b.v,
      /*c_mod=*/(short)0, c, /*reuse_a=*/false, /*reuse_b=*/false);
}

// Async global->LDS copy (GVS form): mem[saddr + voff] -> LDS[lds_addr], 16B.
// Tracked by ASYNCcnt.
static __device__ inline void async_ld_b128(unsigned lds_addr,
                                            unsigned long long sbase,
                                            unsigned voff) {
  asm volatile("global_load_async_to_lds_b128 %0, %1, %2"
               :: "v"(lds_addr), "v"(voff), "s"(sbase)
               : "memory");
}

static __device__ inline void wait_async0() {
  asm volatile("s_wait_asynccnt 0x0" ::: "memory");
}

// ---------------------------------------------------------------------------
// fp32 -> bf16 conversion
// ---------------------------------------------------------------------------
__global__ __launch_bounds__(256) void f32_to_bf16(const float* __restrict__ x,
                                                   __bf16* __restrict__ y, int n) {
  const int i = blockIdx.x * blockDim.x + threadIdx.x;
  if (i < n) y[i] = (__bf16)x[i];
}

// ---------------------------------------------------------------------------
// Generic bf16 WMMA GEMM: out[m,n] = sum_k A[m,k] * W[n,k]
// A: [M,K] row-major bf16, W: [N,K] row-major bf16 (torch Linear layout).
// Block = 256 threads = 8 waves; block tile 128(M) x 64(N); wave tile 16 x 64.
// The W tile (shared by all 8 waves) is staged in LDS with double-buffered
// GLOBAL_LOAD_ASYNC_TO_LDS_B128 (one 16B chunk per thread per K-step).
// ---------------------------------------------------------------------------
template <bool OUT_F32>
__global__ __launch_bounds__(256) void gemm_bf16_wmma(
    const __bf16* __restrict__ A, const __bf16* __restrict__ W,
    void* __restrict__ out, int M, int N, int K) {
  __shared__ __align__(16) __bf16 smW[2][64 * 32];  // 2 x 4KB W tiles [n][k]

  const int tid  = threadIdx.x;
  const int lane = tid & 31;
  const int wave = tid >> 5;
  const int mrow = lane & 15;  // A-frag row / B-frag column / C-frag column
  const int g    = lane >> 4;  // half-wave group
  const int m0 = blockIdx.y * 128 + wave * 16;
  const int n0 = blockIdx.x * 64;

  // Per-thread async-copy slot: row wn in [0,64), 8-elem k chunk wk.
  const int wn = tid >> 2;
  const int wk = (tid & 3) * 8;
  const unsigned lds0   = (unsigned)(uintptr_t)(&smW[0][0]);
  const unsigned ldsoff = (unsigned)((wn * 32 + wk) * 2);
  const unsigned voff   = (unsigned)((wn * K + wk) * 2);
  const unsigned long long wbase =
      (unsigned long long)(uintptr_t)(W + (size_t)n0 * K);

  const __bf16* Arow = A + (size_t)(m0 + mrow) * K;

  v8f acc[4] = {vzero(), vzero(), vzero(), vzero()};

  // Preload tile 0.
  async_ld_b128(lds0 + ldsoff, wbase, voff);

  const int nk = K / 32;
  for (int i = 0; i < nk; ++i) {
    wait_async0();    // my chunk of tile i landed
    __syncthreads();  // everyone's chunks landed; prior compute done
    if (i + 1 < nk)   // stage tile i+1 into the other buffer (k advances 64B)
      async_ld_b128(lds0 + ((unsigned)((i + 1) & 1)) * 4096u + ldsoff,
                    wbase + (unsigned long long)(i + 1) * 64ull, voff);

    const int k0 = i * 32;
    Frag a;  // A 16x32 layout: elems 0-15 -> k0+16g.. (contiguous)
    a.h[0] = *(const v8bf*)(Arow + k0 + g * 16);
    a.h[1] = *(const v8bf*)(Arow + k0 + g * 16 + 8);
    __builtin_prefetch(Arow + k0 + 64, 0, 1);

    const __bf16* bt = &smW[i & 1][0] + g * 16;
#pragma unroll
    for (int nt = 0; nt < 4; ++nt) {
      Frag b;  // B 32x16 layout: col = lane&15, k = j + 16g (contiguous in LDS)
      const __bf16* p = bt + (nt * 16 + mrow) * 32;
      b.h[0] = *(const v8bf*)(p);
      b.h[1] = *(const v8bf*)(p + 8);
      acc[nt] = wmma_bf16(a, b, acc[nt]);
    }
  }

#pragma unroll
  for (int nt = 0; nt < 4; ++nt) {
    const int col = n0 + nt * 16 + mrow;
#pragma unroll
    for (int r = 0; r < 8; ++r) {  // C layout: elem r -> row r+8g, col lane&15
      const size_t row = (size_t)(m0 + r + 8 * g);
      const float val = acc[nt][r];
      if constexpr (OUT_F32)
        ((float*)out)[row * N + col] = val;
      else
        ((__bf16*)out)[row * N + col] = (__bf16)val;
    }
  }
}

// ---------------------------------------------------------------------------
// RoPE + layout reorder: qkv[m, 3C] bf16 ->
//   Q [B,H,T,D] rotated, K [B,H,T,D] rotated, V^T [B,H,D,T]
// One thread per (b,t,h,i), i in [0,32) handles rotation pair (i, i+32).
// ---------------------------------------------------------------------------
__global__ __launch_bounds__(256) void rope_reorder(
    const __bf16* __restrict__ qkv, __bf16* __restrict__ Qo,
    __bf16* __restrict__ Ko, __bf16* __restrict__ Vt) {
  const int idx = blockIdx.x * blockDim.x + threadIdx.x;  // 2^22 total
  const int i = idx & 31;
  const int h = (idx >> 5) & 15;
  const int t = (idx >> 9) & 2047;
  const int b = idx >> 20;

  const __bf16* row = qkv + (size_t)(b * T_ + t) * (3 * C_) + h * D_;
  const float q1 = (float)row[i],      q2 = (float)row[i + 32];
  const float k1 = (float)row[C_ + i], k2 = (float)row[C_ + i + 32];

  // inv_freq = 10000^(-2i/64) = exp(-2i * ln(10000)/64)
  const float freq = __expf((float)(-2 * i) * 0.14391156f);
  const float ang  = (float)t * freq;
  const float c = __cosf(ang), s = __sinf(ang);

  const size_t hd = ((size_t)(b * H_ + h) * T_ + t) * D_;
  Qo[hd + i]      = (__bf16)(q1 * c - q2 * s);
  Qo[hd + i + 32] = (__bf16)(q2 * c + q1 * s);
  Ko[hd + i]      = (__bf16)(k1 * c - k2 * s);
  Ko[hd + i + 32] = (__bf16)(k2 * c + k1 * s);

  const size_t vt = (size_t)(b * H_ + h) * D_ * T_ + t;
  Vt[vt + (size_t)i * T_]        = row[2 * C_ + i];
  Vt[vt + (size_t)(i + 32) * T_] = row[2 * C_ + i + 32];
}

// ---------------------------------------------------------------------------
// Flash attention: one wave per 16-row q-tile, 32-key blocks, online softmax.
// Q,K: [B,H,T,D] bf16 (rotated). Vt: [B,H,D,T] bf16. O: [B,T,C] bf16.
// ---------------------------------------------------------------------------
__global__ __launch_bounds__(128) void flash_attn(
    const __bf16* __restrict__ Q, const __bf16* __restrict__ Kt,
    const __bf16* __restrict__ Vt, __bf16* __restrict__ O) {
  __shared__ __align__(16) __bf16 smem[4][16 * 32];  // per-wave P staging tile

  const int lane = threadIdx.x & 31;
  const int wave = threadIdx.x >> 5;
  const int mrow = lane & 15;
  const int g    = lane >> 4;

  const int idx = blockIdx.x * 4 + wave;    // [0, B*H*T/16)
  const int qt = idx & (T_ / 16 - 1);
  const int h  = (idx >> 7) & (H_ - 1);
  const int b  = idx >> 11;
  const int q0 = qt * 16;

  const __bf16* Qb = Q  + ((size_t)(b * H_ + h) * T_ + q0) * D_;
  const __bf16* Kb = Kt + (size_t)(b * H_ + h) * T_ * D_;
  const __bf16* Vb = Vt + (size_t)(b * H_ + h) * D_ * T_;

  // Q A-fragments for d=[0,32) and d=[32,64)
  Frag aq0, aq1;
  {
    const __bf16* qp = Qb + (size_t)mrow * D_ + g * 16;
    aq0.h[0] = *(const v8bf*)(qp);
    aq0.h[1] = *(const v8bf*)(qp + 8);
    aq1.h[0] = *(const v8bf*)(qp + 32);
    aq1.h[1] = *(const v8bf*)(qp + 40);
  }

  float mR[8], lR[8];
#pragma unroll
  for (int r = 0; r < 8; ++r) { mR[r] = -3.0e38f; lR[r] = 0.f; }
  v8f o[4] = {vzero(), vzero(), vzero(), vzero()};

  const int nb = (q0 + 47) >> 5;  // ceil((q0+16)/32) key blocks
  for (int kb = 0; kb < nb; ++kb) {
    const int kpos = kb * 32;

    // K B-fragments: B[k,n] = K[key=n, d=k]  (row-major [T,D], contiguous in k)
    const __bf16* kp0 = Kb + (size_t)(kpos + mrow) * D_ + g * 16;
    const __bf16* kp1 = kp0 + 16 * D_;
    Frag b00, b01, b10, b11;
    b00.h[0] = *(const v8bf*)(kp0);      b00.h[1] = *(const v8bf*)(kp0 + 8);
    b01.h[0] = *(const v8bf*)(kp0 + 32); b01.h[1] = *(const v8bf*)(kp0 + 40);
    b10.h[0] = *(const v8bf*)(kp1);      b10.h[1] = *(const v8bf*)(kp1 + 8);
    b11.h[0] = *(const v8bf*)(kp1 + 32); b11.h[1] = *(const v8bf*)(kp1 + 40);
    __builtin_prefetch(kp0 + 32 * D_, 0, 1);

    v8f s0 = vzero(), s1 = vzero();
    s0 = wmma_bf16(aq0, b00, s0);
    s0 = wmma_bf16(aq1, b01, s0);
    s1 = wmma_bf16(aq0, b10, s1);
    s1 = wmma_bf16(aq1, b11, s1);

    const float sc = 0.125f;  // 1/sqrt(64)
#pragma unroll
    for (int r = 0; r < 8; ++r) { s0[r] *= sc; s1[r] *= sc; }

    if (kb == nb - 1) {  // causal mask on the diagonal block
      const int c0 = kpos + mrow, c1 = c0 + 16;
#pragma unroll
      for (int r = 0; r < 8; ++r) {
        const int qr = q0 + r + 8 * g;
        if (c0 > qr) s0[r] = -3.0e38f;
        if (c1 > qr) s1[r] = -3.0e38f;
      }
    }

    // Online softmax. Row r+8g lives at element r across the 16 lanes of half g.
    float mx[8], p0[8], p1[8], sf[8], rs[8];
#pragma unroll
    for (int r = 0; r < 8; ++r) mx[r] = fmaxf(s0[r], s1[r]);
#pragma unroll
    for (int off = 8; off >= 1; off >>= 1)
#pragma unroll
      for (int r = 0; r < 8; ++r)
        mx[r] = fmaxf(mx[r], __shfl_xor(mx[r], off, 16));
#pragma unroll
    for (int r = 0; r < 8; ++r) {
      const float mn = fmaxf(mR[r], mx[r]);
      sf[r] = __expf(mR[r] - mn);
      mR[r] = mn;
      p0[r] = __expf(s0[r] - mn);
      p1[r] = __expf(s1[r] - mn);
      rs[r] = p0[r] + p1[r];
    }
#pragma unroll
    for (int off = 8; off >= 1; off >>= 1)
#pragma unroll
      for (int r = 0; r < 8; ++r)
        rs[r] += __shfl_xor(rs[r], off, 16);
#pragma unroll
    for (int r = 0; r < 8; ++r) {
      lR[r] = lR[r] * sf[r] + rs[r];
#pragma unroll
      for (int dt = 0; dt < 4; ++dt) o[dt][r] *= sf[r];
    }

    // P (C layout, f32) -> LDS (16x32 bf16, row-major) -> A-fragment.
    // Wave-private region; same-wave DS ops are in-order (DScnt).
    __bf16* pl = &smem[wave][0];
#pragma unroll
    for (int r = 0; r < 8; ++r) {
      pl[(r + 8 * g) * 32 + mrow]      = (__bf16)p0[r];
      pl[(r + 8 * g) * 32 + 16 + mrow] = (__bf16)p1[r];
    }
    Frag ap;
    ap.h[0] = *(const v8bf*)(pl + mrow * 32 + g * 16);
    ap.h[1] = *(const v8bf*)(pl + mrow * 32 + g * 16 + 8);

    // V B-fragments: B[k,n] = Vt[d=n, key=k]  ([D,T], contiguous in k)
    const __bf16* vp = Vb + (size_t)mrow * T_ + kpos + g * 16;
#pragma unroll
    for (int dt = 0; dt < 4; ++dt) {
      Frag bv;
      const __bf16* p = vp + (size_t)(dt * 16) * T_;
      bv.h[0] = *(const v8bf*)(p);
      bv.h[1] = *(const v8bf*)(p + 8);
      o[dt] = wmma_bf16(ap, bv, o[dt]);
    }
  }

  // Normalize and write attention output in [B,T,C] bf16 for the proj GEMM.
#pragma unroll
  for (int r = 0; r < 8; ++r) {
    const float inv = 1.0f / lR[r];
    const size_t row = (size_t)(b * T_ + q0 + r + 8 * g) * C_ + h * D_;
#pragma unroll
    for (int dt = 0; dt < 4; ++dt)
      O[row + dt * 16 + mrow] = (__bf16)(o[dt][r] * inv);
  }
}

// ---------------------------------------------------------------------------
// Launch
// ---------------------------------------------------------------------------
extern "C" void kernel_launch(void* const* d_in, const int* in_sizes, int n_in,
                              void* d_out, int out_size, void* d_ws, size_t ws_size,
                              hipStream_t stream) {
  const float* x      = (const float*)d_in[0];  // [B,T,C]
  const float* w_qkv  = (const float*)d_in[1];  // [3C,C]
  const float* w_proj = (const float*)d_in[2];  // [C,C]
  float* out = (float*)d_out;                   // [B,T,C] f32

  const size_t MT = (size_t)B_ * T_;  // 8192 rows
  __bf16* ws = (__bf16*)d_ws;
  size_t off = 0;
  __bf16* xb     = ws + off; off += MT * C_;            // x bf16
  __bf16* wqkvb  = ws + off; off += (size_t)3 * C_ * C_;
  __bf16* wprojb = ws + off; off += (size_t)C_ * C_;
  __bf16* qkvb   = ws + off; off += MT * 3 * C_;        // qkv (pre-RoPE)
  __bf16* Qb     = ws + off; off += MT * C_;            // [B,H,T,D]
  __bf16* Kb     = ws + off; off += MT * C_;            // [B,H,T,D]
  __bf16* Vtb    = ws + off; off += MT * C_;            // [B,H,D,T]
  __bf16* attnob = qkvb;  // reuse: qkv dead after rope_reorder

  // 1) convert inputs to bf16
  f32_to_bf16<<<(int)((MT * C_) / 256), 256, 0, stream>>>(x, xb, (int)(MT * C_));
  f32_to_bf16<<<(3 * C_ * C_) / 256, 256, 0, stream>>>(w_qkv, wqkvb, 3 * C_ * C_);
  f32_to_bf16<<<(C_ * C_) / 256, 256, 0, stream>>>(w_proj, wprojb, C_ * C_);

  // 2) QKV GEMM: [8192,1024] x [3072,1024]^T -> [8192,3072] bf16
  gemm_bf16_wmma<false><<<dim3((3 * C_) / 64, MT / 128), 256, 0, stream>>>(
      xb, wqkvb, qkvb, (int)MT, 3 * C_, C_);

  // 3) RoPE + reorder into Q,K [B,H,T,D] and V^T [B,H,D,T]
  rope_reorder<<<(B_ * T_ * H_ * 32) / 256, 256, 0, stream>>>(qkvb, Qb, Kb, Vtb);

  // 4) causal flash attention -> [B,T,C] bf16
  flash_attn<<<(B_ * H_ * (T_ / 16)) / 4, 128, 0, stream>>>(Qb, Kb, Vtb, attnob);

  // 5) output projection: [8192,1024] x [1024,1024]^T -> d_out f32
  gemm_bf16_wmma<true><<<dim3(C_ / 64, MT / 128), 256, 0, stream>>>(
      attnob, wprojb, out, (int)MT, C_, C_);
}
